// GATNet_Object_25262997635276
// MI455X (gfx1250) — matrix operand order repaired and checked by reading, not compiled, not run on hardware
//
#include <hip/hip_runtime.h>
#include <hip/hip_bf16.h>
#include <cstdint>
#include <cstddef>

#define N_NODES 20000
#define MP      20032                 // M padded to 64-row block tiles (313 * 64)
#define N_EDGES 320000
#define EE_TOT  (N_EDGES + N_NODES)   // edges + self loops
#define DIN     4096
#define EMBD    200
#define HIDDEN  512
#define NHEAD   8
#define HC      64
#define NCLS    151
#define D1      4424                  // 4096 + 200 + 128
#define D1P     4448                  // padded to multiple of 32 for WMMA K
#define REPD    4808                  // 4096 + 512 + 200
#define NEG_SLOPE 0.2f
#define LDSROW  40                    // A-tile LDS row stride in bf16 (80B, bank-spread)

typedef __bf16 v16bf __attribute__((ext_vector_type(16)));
typedef float  v8f   __attribute__((ext_vector_type(8)));

static __device__ inline __bf16 f2bf(float f) {
  union { float f; unsigned u; } x; x.f = f;
  unsigned r = (x.u + 0x7FFFu + ((x.u >> 16) & 1u)) >> 16;
  unsigned short h = (unsigned short)r;
  return __builtin_bit_cast(__bf16, h);
}

static __device__ inline void atomicMaxF(float* addr, float val) {
  int* ia = (int*)addr;
  int old = __float_as_int(*addr);
  while (__int_as_float(old) < val) {
    int assumed = old;
    old = atomicCAS(ia, assumed, __float_as_int(val));
    if (old == assumed) break;
  }
}

// ---------------- input feature assembly: x = [roi | emb1[lab] | bboxMLP] (bf16, K padded) ----
__global__ __launch_bounds__(128)
void build_x_kernel(const float* __restrict__ roi, const float* __restrict__ box,
                    const float* __restrict__ emb1,
                    const float* __restrict__ bw1, const float* __restrict__ bb1,
                    const float* __restrict__ bw2, const float* __restrict__ bb2,
                    const int* __restrict__ labels, __bf16* __restrict__ xb) {
  __shared__ float sb[9];
  __shared__ float hid[32];
  __shared__ float pos[128];
  const int n = blockIdx.x;
  const int t = threadIdx.x;
  if (t < 9) sb[t] = box[n * 9 + t];
  __syncthreads();
  if (t < 32) {
    float a = bb1[t];
    for (int i = 0; i < 9; ++i) a += sb[i] * bw1[i * 32 + t];
    hid[t] = a > 0.f ? a : 0.f;
  }
  __syncthreads();
  {
    float a = bb2[t];
    for (int i = 0; i < 32; ++i) a += hid[i] * bw2[i * 128 + t];
    pos[t] = a > 0.f ? a : 0.f;
  }
  __syncthreads();
  const int lab = labels[n];
  __bf16* xrow = xb + (size_t)n * D1P;
  for (int j = t; j < D1P; j += 128) {
    float v;
    if (j < DIN)             v = roi[(size_t)n * DIN + j];
    else if (j < DIN + EMBD) v = emb1[lab * EMBD + (j - DIN)];
    else if (j < D1)         v = pos[j - (DIN + EMBD)];
    else                     v = 0.f;
    xrow[j] = f2bf(v);
  }
}

// ---------------- W[k][n] (f32) -> Wt[n][k] (bf16, k padded with zeros) ------------------------
__global__ void transpose_to_bf16_kernel(const float* __restrict__ W, __bf16* __restrict__ Wt,
                                         int K, int Kp, int ncol, long total) {
  long id = (long)blockIdx.x * blockDim.x + threadIdx.x;
  if (id >= total) return;
  int nc = (int)(id / Kp);
  int k  = (int)(id % Kp);
  Wt[(size_t)nc * Kp + k] = (k < K) ? f2bf(W[(size_t)k * ncol + nc]) : f2bf(0.f);
}

__global__ void fill_kernel(float* __restrict__ p, float v, long n) {
  long i = (long)blockIdx.x * blockDim.x + threadIdx.x;
  if (i < n) p[i] = v;
}

__global__ void fill_u32_kernel(unsigned* __restrict__ p, unsigned v, long n) {
  long i = (long)blockIdx.x * blockDim.x + threadIdx.x;
  if (i < n) p[i] = v;
}

// ---------------- bf16 WMMA GEMM: C[MP,512] = A[MP,K] * Bt[512,K]^T, f32 accumulate ------------
// Block = 64 M rows x 512 N cols, 8 waves; each wave owns a 64x64 tile (4x4 WMMA accumulators).
// A tile (64x32 bf16) is double-buffered through LDS and shared by all waves; B streams from
// global (L2-resident). 16 WMMAs per 32-K step per wave.
union FragBF { v16bf v; uint4 u[2]; };

__global__ __launch_bounds__(256)
void gemm_bf16_v2(const __bf16* __restrict__ A, const __bf16* __restrict__ Bt,
                  float* __restrict__ C, int K) {
  __shared__ __bf16 sA[2][64 * LDSROW];
  const int lane = threadIdx.x & 31;
  const int wave = threadIdx.x >> 5;
  const int m0   = blockIdx.x * 64;
  const int n0   = wave * 64;
  const int aSel = (lane < 16) ? 0 : 8;    // 16-bit A layout: K 0..7/16..23 vs 8..15/24..31
  const int bSel = (lane < 16) ? 0 : 8;    // 16-bit B layout: K 0..7/16..23 vs 8..15/24..31
  const int l15  = lane & 15;

  // cooperative A-tile load mapping: 256 threads x 16B = 64 rows x 32 bf16
  const int lrow = threadIdx.x >> 2;
  const int lk   = (threadIdx.x & 3) * 8;
  const __bf16* agp = A + (size_t)(m0 + lrow) * K + lk;

  v8f acc[4][4] = {};

  // preload k0 = 0 tile
  {
    uint4 v = *(const uint4*)agp;
    *(uint4*)&sA[0][lrow * LDSROW + lk] = v;
  }
  __syncthreads();

  const int iters = K >> 5;
  for (int it = 0; it < iters; ++it) {
    const int buf = it & 1;
    const bool have_next = (it + 1) < iters;
    uint4 nexta;
    if (have_next) nexta = *(const uint4*)(agp + (it + 1) * 32);

    // A fragments from LDS (shared across waves)
    FragBF a[4];
#pragma unroll
    for (int ms = 0; ms < 4; ++ms) {
      const __bf16* ap = &sA[buf][(ms * 16 + l15) * LDSROW + aSel];
      a[ms].u[0] = *(const uint4*)ap;
      a[ms].u[1] = *(const uint4*)(ap + 16);
    }
    // B fragments from global (L2-resident weight panel)
    FragBF b[4];
    const int k0 = it * 32;
#pragma unroll
    for (int ns = 0; ns < 4; ++ns) {
      const __bf16* bp = Bt + (size_t)(n0 + ns * 16 + l15) * K + k0 + bSel;
      b[ns].u[0] = *(const uint4*)bp;
      b[ns].u[1] = *(const uint4*)(bp + 16);
    }
#pragma unroll
    for (int ms = 0; ms < 4; ++ms)
#pragma unroll
      for (int ns = 0; ns < 4; ++ns)
        acc[ms][ns] = __builtin_amdgcn_wmma_f32_16x16x32_bf16(
            false, a[ms].v, false, b[ns].v, (short)0, acc[ms][ns], false, false);

    if (have_next) *(uint4*)&sA[buf ^ 1][lrow * LDSROW + lk] = nexta;
    __syncthreads();
  }

  const int crow0 = ((lane < 16) ? 0 : 8);
#pragma unroll
  for (int ms = 0; ms < 4; ++ms) {
#pragma unroll
    for (int ns = 0; ns < 4; ++ns) {
      float* cp = C + (size_t)(m0 + ms * 16 + crow0) * HIDDEN + n0 + ns * 16 + l15;
#pragma unroll
      for (int r = 0; r < 8; ++r)
        cp[(size_t)r * HIDDEN] = acc[ms][ns][r];
    }
  }
}

// ---------------- attention source/dest dots per node/head -------------------------------------
__global__ void att_dots_kernel(const float* __restrict__ h, const float* __restrict__ a_src,
                                const float* __restrict__ a_dst, float* __restrict__ als,
                                float* __restrict__ ald, int H, int C) {
  int id = blockIdx.x * blockDim.x + threadIdx.x;
  if (id >= N_NODES * H) return;
  int n = id / H, hh = id % H;
  const float* hp = h + (size_t)n * H * C + (size_t)hh * C;
  const float* as = a_src + hh * C;
  const float* ad = a_dst + hh * C;
  float s = 0.f, d = 0.f;
  for (int c = 0; c < C; ++c) { float v = hp[c]; s += v * as[c]; d += v * ad[c]; }
  als[id] = s; ald[id] = d;
}

static __device__ inline void edge_src_dst(const int* __restrict__ ei, int e, int& s, int& d) {
  if (e < N_EDGES) { s = ei[e]; d = ei[N_EDGES + e]; }
  else             { s = d = e - N_EDGES; }
}

// ---------------- segment softmax pass 1: leaky logits + per-dst max ---------------------------
__global__ void edge_logits_max_kernel(const float* __restrict__ als, const float* __restrict__ ald,
                                       const int* __restrict__ ei, float* __restrict__ elog,
                                       float* __restrict__ m, int H) {
  long id = (long)blockIdx.x * blockDim.x + threadIdx.x;
  if (id >= (long)EE_TOT * H) return;
  int e = (int)(id / H), hh = (int)(id % H);
  int s, d; edge_src_dst(ei, e, s, d);
  float x = als[s * H + hh] + ald[d * H + hh];
  x = x > 0.f ? x : NEG_SLOPE * x;
  elog[id] = x;
  atomicMaxF(&m[d * H + hh], x);
}

// ---------------- pass 2: exp(logit - max) + per-dst sum ---------------------------------------
__global__ void edge_exp_sum_kernel(const int* __restrict__ ei, float* __restrict__ elog,
                                    const float* __restrict__ m, float* __restrict__ ssum, int H) {
  long id = (long)blockIdx.x * blockDim.x + threadIdx.x;
  if (id >= (long)EE_TOT * H) return;
  int e = (int)(id / H), hh = (int)(id % H);
  int s, d; edge_src_dst(ei, e, s, d);
  float ev = expf(elog[id] - m[d * H + hh]);
  elog[id] = ev;
  atomicAdd(&ssum[d * H + hh], ev);
}

// ---------------- pass 3: out[dst] += alpha * h[src], 64 channels per thread -------------------
__global__ void edge_aggregate_kernel(const int* __restrict__ ei, const float* __restrict__ elog,
                                      const float* __restrict__ ssum, const float* __restrict__ h,
                                      float* __restrict__ out, int H, int C) {
  const int chunks = C >> 6;
  long id = (long)blockIdx.x * blockDim.x + threadIdx.x;
  long tot = (long)EE_TOT * H * chunks;
  if (id >= tot) return;
  int cc  = (int)(id % chunks);
  long r  = id / chunks;
  int hh  = (int)(r % H);
  int e   = (int)(r / H);
  int s, d; edge_src_dst(ei, e, s, d);
  float alpha = elog[(long)e * H + hh] / (ssum[d * H + hh] + 1e-16f);
  const float* hp = h + (size_t)s * H * C + (size_t)hh * C + cc * 64;
  float* op       = out + (size_t)d * H * C + (size_t)hh * C + cc * 64;
  for (int c = 0; c < 64; ++c) atomicAdd(&op[c], alpha * hp[c]);
}

// ---------------- bias + ELU, emit bf16 for GEMM2 ----------------------------------------------
__global__ void bias_elu_bf16_kernel(const float* __restrict__ in, const float* __restrict__ bias,
                                     __bf16* __restrict__ out, long n) {
  long i = (long)blockIdx.x * blockDim.x + threadIdx.x;
  if (i >= n) return;
  float v = in[i] + bias[i % HIDDEN];
  v = v > 0.f ? v : (expf(v) - 1.f);
  out[i] = f2bf(v);
}

// ---------------- final outputs ----------------------------------------------------------------
__global__ void onehot_kernel(const int* __restrict__ labels, float* __restrict__ dout) {
  long id = (long)blockIdx.x * blockDim.x + threadIdx.x;
  if (id >= (long)N_NODES * NCLS) return;
  int n = (int)(id / NCLS), c = (int)(id % NCLS);
  dout[id] = (labels[n] == c) ? 1.f : 0.f;
}

__global__ void labelsf_kernel(const int* __restrict__ labels, float* __restrict__ dout) {
  int n = blockIdx.x * blockDim.x + threadIdx.x;
  if (n >= N_NODES) return;
  dout[(long)N_NODES * NCLS + n] = (float)labels[n];
}

__global__ void rep_kernel(const float* __restrict__ roi, const float* __restrict__ emb2,
                           const float* __restrict__ out2, const float* __restrict__ bias2,
                           const int* __restrict__ labels, float* __restrict__ dout) {
  long id = (long)blockIdx.x * blockDim.x + threadIdx.x;
  if (id >= (long)N_NODES * REPD) return;
  int n = (int)(id / REPD), j = (int)(id % REPD);
  float v;
  if (j < DIN)               v = roi[(size_t)n * DIN + j];
  else if (j < DIN + HIDDEN) v = out2[(size_t)n * HIDDEN + (j - DIN)] + bias2[j - DIN];
  else                       v = emb2[labels[n] * EMBD + (j - DIN - HIDDEN)];
  dout[(long)N_NODES * NCLS + N_NODES + id] = v;
}

// ================================================================================================
extern "C" void kernel_launch(void* const* d_in, const int* in_sizes, int n_in,
                              void* d_out, int out_size, void* d_ws, size_t ws_size,
                              hipStream_t stream) {
  const float* roi    = (const float*)d_in[0];
  const float* box    = (const float*)d_in[1];
  const float* emb1   = (const float*)d_in[2];
  const float* emb2   = (const float*)d_in[3];
  const float* bw1    = (const float*)d_in[4];
  const float* bb1    = (const float*)d_in[5];
  const float* bw2    = (const float*)d_in[6];
  const float* bb2    = (const float*)d_in[7];
  const float* W1     = (const float*)d_in[8];
  const float* a_src1 = (const float*)d_in[9];
  const float* a_dst1 = (const float*)d_in[10];
  const float* bias1  = (const float*)d_in[11];
  const float* W2     = (const float*)d_in[12];
  const float* a_src2 = (const float*)d_in[13];
  const float* a_dst2 = (const float*)d_in[14];
  const float* bias2  = (const float*)d_in[15];
  const int*   labels = (const int*)d_in[16];
  const int*   ei     = (const int*)d_in[17];
  float* dout = (float*)d_out;

  // ---- workspace carve-out (M-padded matrices for the 64-row block tiles) ----
  char* ws = (char*)d_ws;
  size_t off = 0;
  auto take = [&](size_t bytes) -> char* {
    off = (off + 255) & ~(size_t)255;
    char* p = ws + off; off += bytes; return p;
  };
  __bf16* xb   = (__bf16*)take((size_t)MP * D1P * 2);
  __bf16* w1t  = (__bf16*)take((size_t)HIDDEN * D1P * 2);
  float*  h1   = (float*) take((size_t)MP * HIDDEN * 4);
  float*  als1 = (float*) take((size_t)N_NODES * NHEAD * 4);
  float*  ald1 = (float*) take((size_t)N_NODES * NHEAD * 4);
  float*  m1   = (float*) take((size_t)N_NODES * NHEAD * 4);
  float*  s1   = (float*) take((size_t)N_NODES * NHEAD * 4);
  float*  el1  = (float*) take((size_t)EE_TOT * NHEAD * 4);
  float*  out1 = (float*) take((size_t)N_NODES * HIDDEN * 4);
  __bf16* e1b  = (__bf16*)take((size_t)MP * HIDDEN * 2);
  __bf16* w2t  = (__bf16*)take((size_t)HIDDEN * HIDDEN * 2);
  float*  h2   = (float*) take((size_t)MP * HIDDEN * 4);
  float*  als2 = (float*) take((size_t)N_NODES * 4);
  float*  ald2 = (float*) take((size_t)N_NODES * 4);
  float*  m2   = (float*) take((size_t)N_NODES * 4);
  float*  s2   = (float*) take((size_t)N_NODES * 4);
  float*  el2  = (float*) take((size_t)EE_TOT * 4);
  float*  out2 = (float*) take((size_t)N_NODES * HIDDEN * 4);

  auto blocks = [](long n) { return dim3((unsigned)((n + 255) / 256)); };

  // 1) assemble x in bf16 (padded); zero the M pad rows
  build_x_kernel<<<dim3(N_NODES), dim3(128), 0, stream>>>(roi, box, emb1, bw1, bb1, bw2, bb2,
                                                          labels, xb);
  {
    long padw = (long)(MP - N_NODES) * D1P / 2;   // bf16 pairs as u32
    fill_u32_kernel<<<blocks(padw), dim3(256), 0, stream>>>(
        (unsigned*)(xb + (size_t)N_NODES * D1P), 0u, padw);
  }
  // 2) W1 -> bf16 transposed [512][4448]
  {
    long tot = (long)HIDDEN * D1P;
    transpose_to_bf16_kernel<<<blocks(tot), dim3(256), 0, stream>>>(W1, w1t, D1, D1P, HIDDEN, tot);
  }
  // 3) init accumulators
  fill_kernel<<<blocks((long)N_NODES * NHEAD), dim3(256), 0, stream>>>(m1, -3.4e38f, (long)N_NODES * NHEAD);
  fill_kernel<<<blocks((long)N_NODES * NHEAD), dim3(256), 0, stream>>>(s1, 0.f, (long)N_NODES * NHEAD);
  fill_kernel<<<blocks((long)N_NODES * HIDDEN), dim3(256), 0, stream>>>(out1, 0.f, (long)N_NODES * HIDDEN);
  fill_kernel<<<blocks((long)N_NODES), dim3(256), 0, stream>>>(m2, -3.4e38f, (long)N_NODES);
  fill_kernel<<<blocks((long)N_NODES), dim3(256), 0, stream>>>(s2, 0.f, (long)N_NODES);
  fill_kernel<<<blocks((long)N_NODES * HIDDEN), dim3(256), 0, stream>>>(out2, 0.f, (long)N_NODES * HIDDEN);

  // 4) GEMM1: h1 = x @ W1  (20032x4448 * 4448x512, bf16 WMMA, f32 acc, LDS-staged A)
  gemm_bf16_v2<<<dim3(MP / 64), dim3(256), 0, stream>>>(xb, w1t, h1, D1P);

  // 5) GAT layer 1 attention
  att_dots_kernel<<<blocks((long)N_NODES * NHEAD), dim3(256), 0, stream>>>(h1, a_src1, a_dst1,
                                                                           als1, ald1, NHEAD, HC);
  edge_logits_max_kernel<<<blocks((long)EE_TOT * NHEAD), dim3(256), 0, stream>>>(als1, ald1, ei,
                                                                                 el1, m1, NHEAD);
  edge_exp_sum_kernel<<<blocks((long)EE_TOT * NHEAD), dim3(256), 0, stream>>>(ei, el1, m1, s1, NHEAD);
  edge_aggregate_kernel<<<blocks((long)EE_TOT * NHEAD), dim3(256), 0, stream>>>(ei, el1, s1, h1,
                                                                                out1, NHEAD, HC);
  // 6) bias + ELU -> bf16 input for layer 2 (zero the pad rows)
  bias_elu_bf16_kernel<<<blocks((long)N_NODES * HIDDEN), dim3(256), 0, stream>>>(
      out1, bias1, e1b, (long)N_NODES * HIDDEN);
  {
    long padw = (long)(MP - N_NODES) * HIDDEN / 2;
    fill_u32_kernel<<<blocks(padw), dim3(256), 0, stream>>>(
        (unsigned*)(e1b + (size_t)N_NODES * HIDDEN), 0u, padw);
  }
  // 7) W2 -> bf16 transposed [512][512]
  {
    long tot = (long)HIDDEN * HIDDEN;
    transpose_to_bf16_kernel<<<blocks(tot), dim3(256), 0, stream>>>(W2, w2t, HIDDEN, HIDDEN, HIDDEN, tot);
  }
  // 8) GEMM2: h2 = edge_f1 @ W2
  gemm_bf16_v2<<<dim3(MP / 64), dim3(256), 0, stream>>>(e1b, w2t, h2, HIDDEN);

  // 9) GAT layer 2 attention (1 head, 512 channels)
  att_dots_kernel<<<blocks((long)N_NODES), dim3(256), 0, stream>>>(h2, a_src2, a_dst2,
                                                                   als2, ald2, 1, HIDDEN);
  edge_logits_max_kernel<<<blocks((long)EE_TOT), dim3(256), 0, stream>>>(als2, ald2, ei, el2, m2, 1);
  edge_exp_sum_kernel<<<blocks((long)EE_TOT), dim3(256), 0, stream>>>(ei, el2, m2, s2, 1);
  edge_aggregate_kernel<<<blocks((long)EE_TOT * (HIDDEN / 64)), dim3(256), 0, stream>>>(
      ei, el2, s2, h2, out2, 1, HIDDEN);

  // 10) outputs: one-hot dists, labels-as-float, edge_pre_rep concat (bias2 fused)
  onehot_kernel<<<blocks((long)N_NODES * NCLS), dim3(256), 0, stream>>>(labels, dout);
  labelsf_kernel<<<blocks((long)N_NODES), dim3(256), 0, stream>>>(labels, dout);
  rep_kernel<<<blocks((long)N_NODES * REPD), dim3(256), 0, stream>>>(roi, emb2, out2, bias2,
                                                                     labels, dout);
}
